// KANLayer_89300960018758
// MI455X (gfx1250) — matrix operand order repaired and checked by reading, not compiled, not run on hardware
//
#include <hip/hip_runtime.h>

typedef __attribute__((ext_vector_type(16))) _Float16 v16h;
typedef __attribute__((ext_vector_type(8)))  _Float16 v8h;
typedef __attribute__((ext_vector_type(8)))  float    v8f;

#define IN_F   1024
#define OUT_F  256
#define NB     13          // NUM_BASIS
#define NBP    16          // padded basis per input feature
#define KTOT   (IN_F*NBP)  // 16384 flattened K
#define EPS_K  1e-8f

#define BM 64
#define BN 128
#define BK 32              // K per step = two input features

__device__ __forceinline__ float fast_tanh(float x) {
#if __has_builtin(__builtin_amdgcn_tanhf)
  return __builtin_amdgcn_tanhf(x);      // v_tanh_f32 (gfx1250 trans op)
#else
  return tanhf(x);
#endif
}

// ---------------------------------------------------------------------------
// Kernel 1: coeffs (256,1024,13) f32 -> padded f16 (256,1024,16) in d_ws
// ---------------------------------------------------------------------------
__global__ void kan_convert_coeffs(const float* __restrict__ src,
                                   _Float16* __restrict__ dst, int total) {
  int idx = blockIdx.x * 256 + threadIdx.x;
  if (idx >= total) return;
  int k  = idx & (NBP - 1);
  int ji = idx >> 4;                       // j*IN_F + i
  float v = (k < NB) ? src[ji * NB + k] : 0.0f;
  dst[idx] = (_Float16)v;
}

// ---------------------------------------------------------------------------
// Kernel 2: fused basis + WMMA GEMM.  out[b,j] = sum_K basis[b,K]*coeffsT[K,j]
// ---------------------------------------------------------------------------
__global__ __launch_bounds__(256) void kan_wmma_gemm(
    const float* __restrict__ x,        // (rows, 1024)
    const float* __restrict__ knots,    // (16)
    const _Float16* __restrict__ cf,    // (256, 16384) f16, padded
    float* __restrict__ out)            // (rows, 256)
{
  __shared__ _Float16 ldsA[BM * BK];    // 64x32 halfs = 4 KB

  const int tid   = threadIdx.x;
  const int lane  = tid & 31;
  const int wave  = tid >> 5;
  const int waveM = wave & 1;           // 0..1  -> 32 rows each
  const int waveN = wave >> 1;          // 0..3  -> 32 cols each
  const int blockM = blockIdx.x * BM;
  const int blockN = blockIdx.y * BN;
  const int lane16 = lane & 15;
  const int g      = lane >> 4;

  // Hat basis k: up = (x-l)/(c-l+eps) on [l,c), down = (r-x)/(r-c+eps) on [c,r).
  // The masked sum equals max(0, min(up_line, down_line)) since the two lines
  // cross at the center and go negative outside [l, r).  Precompute
  // slope/intercept so each basis value is 2 FMA + min + max (branch-free).
  float upS[NB - 1], upB[NB - 1], dnS[NB - 1], dnB[NB - 1];
#pragma unroll
  for (int k = 0; k < NB - 1; ++k) {
    float l = knots[k], c = knots[k + 1], r = knots[k + 2];
    float iDL = 1.0f / (c - l + EPS_K);
    float iDR = 1.0f / (r - c + EPS_K);
    upS[k] = iDL;  upB[k] = -l * iDL;    // up   = x*iDL - l*iDL
    dnS[k] = -iDR; dnB[k] =  r * iDR;    // down = -x*iDR + r*iDR
  }

  const v8f zacc = {};
  v8f acc[2][2];
#pragma unroll
  for (int s = 0; s < 2; ++s)
#pragma unroll
    for (int t = 0; t < 2; ++t) acc[s][t] = zacc;

  for (int ks = 0; ks < KTOT / BK; ++ks) {
    // ---- stage A tile: 64 rows x 32 K (two input features), 128 tasks ----
    if (tid < BM * 2) {                 // wave-uniform branch (waves 0..3)
      int row = tid >> 1;
      int ip  = tid & 1;
      int ig  = ks * 2 + ip;            // input feature index
      float xv = x[(size_t)(blockM + row) * IN_F + ig];
      float xn = fast_tanh(xv);
      float xc = fminf(1.0f, fmaxf(-1.0f, xn));
      v16h bv = {};
#pragma unroll
      for (int k = 0; k < NB - 1; ++k) {
        float up = fmaf(xc, upS[k], upB[k]);
        float dn = fmaf(xc, dnS[k], dnB[k]);
        bv[k] = (_Float16)fmaxf(0.0f, fminf(up, dn));
      }
      // bv[12..15] stay zero (reference pads basis with a zero column)
      *(v16h*)(&ldsA[row * BK + ip * 16]) = bv;
    }
    __syncthreads();

    // ---- A fragments from LDS, contiguous K order (k = 16*g + s) ----
    v16h a[2];
#pragma unroll
    for (int s = 0; s < 2; ++s) {
      const _Float16* base = &ldsA[(waveM * 32 + s * 16 + lane16) * BK];
      a[s] = *(const v16h*)(base + g * 16);
    }

    // ---- B fragments straight from global f16 coeffs (L2 resident) ----
    v16h b[2];
#pragma unroll
    for (int t = 0; t < 2; ++t) {
      int col = blockN + waveN * 32 + t * 16 + lane16;
      b[t] = *(const v16h*)(cf + (size_t)col * KTOT + ks * BK + g * 16);
    }

    // ---- 4 WMMAs per wave per K step ----
#pragma unroll
    for (int s = 0; s < 2; ++s)
#pragma unroll
      for (int t = 0; t < 2; ++t)
        acc[s][t] = __builtin_amdgcn_wmma_f32_16x16x32_f16(
            false, a[s], false, b[t], (short)0, acc[s][t], false, false);

    __syncthreads();
  }

  // ---- writeback: 16x16 f32 C/D layout ----
#pragma unroll
  for (int s = 0; s < 2; ++s)
#pragma unroll
    for (int t = 0; t < 2; ++t) {
      int col   = blockN + waveN * 32 + t * 16 + lane16;
      int rbase = blockM + waveM * 32 + s * 16 + g * 8;
#pragma unroll
      for (int v = 0; v < 8; ++v)
        out[(size_t)(rbase + v) * OUT_F + col] = acc[s][t][v];
    }
}

// ---------------------------------------------------------------------------
extern "C" void kernel_launch(void* const* d_in, const int* in_sizes, int n_in,
                              void* d_out, int out_size, void* d_ws, size_t ws_size,
                              hipStream_t stream) {
  const float* x      = (const float*)d_in[0];   // (rows, 1024) f32
  const float* coeffs = (const float*)d_in[1];   // (256, 1024, 13) f32
  const float* knots  = (const float*)d_in[2];   // (16) f32
  float*       out    = (float*)d_out;
  _Float16*    cf     = (_Float16*)d_ws;         // (256, 1024, 16) f16

  const int rows  = in_sizes[0] / IN_F;          // 8192
  const int total = OUT_F * IN_F * NBP;          // 4,194,304

  kan_convert_coeffs<<<(total + 255) / 256, 256, 0, stream>>>(coeffs, cf, total);

  dim3 grid(rows / BM, OUT_F / BN);              // (128, 2)
  kan_wmma_gemm<<<grid, 256, 0, stream>>>(x, knots, cf, out);
}
